// RATTransformer_36361193128444
// MI455X (gfx1250) — compile-verified, matrix operand order checked by reading, not run on hardware
//
#include <hip/hip_runtime.h>
#include <cstdint>
#include <cstddef>

// Problem constants (from reference)
#define B_   4
#define S_   512
#define D_   1600
#define H_   50
#define PH_  32
#define NR_  100
#define MS_  (B_ * S_)   // 2048 rows

typedef unsigned short u16;
typedef __bf16 v16bf __attribute__((ext_vector_type(16)));
typedef float  v8f   __attribute__((ext_vector_type(8)));

enum { EP_F32 = 0, EP_BF16 = 1, EP_GELU_BF16 = 2, EP_HEADS_BF16 = 3 };

union FragAB {
  v16bf v;
  uint4 q[2];
};

__device__ __forceinline__ u16 f2bf(float f) {
  unsigned u = __float_as_uint(f);
  unsigned r = u + 0x7FFFu + ((u >> 16) & 1u);   // round-to-nearest-even
  return (u16)(r >> 16);
}

__device__ __forceinline__ v8f v8f_zero() {
  v8f z = {0.f, 0.f, 0.f, 0.f, 0.f, 0.f, 0.f, 0.f};
  return z;
}

// Async DMA: 16 bytes global -> LDS per enabled lane, tracked by ASYNCcnt.
// LDS flat addresses carry the wave-relative LDS byte offset in the low 32
// bits, so truncating the generic
// pointer yields the VDST operand the instruction expects.
__device__ __forceinline__ void g2lds_b128(u16* lds_dst, const u16* gsrc) {
  unsigned lds_off = (unsigned)(uintptr_t)lds_dst;
  asm volatile("global_load_async_to_lds_b128 %0, %1, off"
               :
               : "v"(lds_off), "v"(gsrc)
               : "memory");
}

__device__ __forceinline__ void wait_async() {
  asm volatile("s_wait_asynccnt 0x0" ::: "memory");
}

// Load a 16x32 bf16 fragment from LDS.
// A-matrix layout (ISA 7.12.2): lanes 0-15 hold rows, elems 0..7 = K(+kg)..,
// elems 8..15 = K(+kg+16); lanes 16-31 use kg=8. B operands are staged
// column-contiguous (Bt[n][k]) so the same loader serves both.
__device__ __forceinline__ FragAB ldfrag(const u16* lds, int row0, int stride, int k0) {
  const unsigned l = threadIdx.x & 31u;
  const int lr = (int)(l & 15u);
  const int hi = (int)(l >> 4);
  const u16* p = lds + (size_t)(row0 + lr) * stride + k0 + hi * 8;
  FragAB f;
  f.q[0] = *reinterpret_cast<const uint4*>(p);
  f.q[1] = *reinterpret_cast<const uint4*>(p + 16);
  return f;
}

__device__ __forceinline__ v8f wmma_bf16(const FragAB& a, const FragAB& b, v8f c) {
  return __builtin_amdgcn_wmma_f32_16x16x32_bf16(false, a.v, false, b.v,
                                                 (short)0, c, false, false);
}

// ---------------------------------------------------------------- casts
__global__ void __launch_bounds__(256) kcast(const float* __restrict__ src,
                                             u16* __restrict__ dst, int n) {
  int i = blockIdx.x * 256 + threadIdx.x;
  if (i < n) dst[i] = f2bf(src[i]);
}

// Wt[n][k] = bf16(W[k][n]) via 32x32 LDS tile (coalesced both sides)
__global__ void __launch_bounds__(256) ktranspose_cast(const float* __restrict__ W,
                                                       u16* __restrict__ Wt,
                                                       int K, int N) {
  __shared__ float tile[32][33];
  int kb = blockIdx.y * 32, nb = blockIdx.x * 32;
  int tx = threadIdx.x & 31, ty = threadIdx.x >> 5;  // 8 row-groups
  for (int i = ty; i < 32; i += 8) {
    int k = kb + i, n = nb + tx;
    tile[i][tx] = (k < K && n < N) ? W[(size_t)k * N + n] : 0.f;
  }
  __syncthreads();
  for (int i = ty; i < 32; i += 8) {
    int n = nb + i, k = kb + tx;
    if (n < N && k < K) Wt[(size_t)n * K + k] = f2bf(tile[tx][i]);
  }
}

// ------------------------------------------------- generic bf16 WMMA GEMM
// C[M,N] = A[M,K] @ B[K,N] + bias, B pre-transposed as Bt[N,K] bf16.
// Block tile 128x128x32, 8 waves (2x4), wave tile 64x32.
// Double-buffered LDS stages filled by async DMA: stage i+1 copies overlap
// stage i WMMAs; the only sync per stage is s_wait_asynccnt 0 + barrier.
// Out-of-range Bt rows are source-CLAMPED instead of zero-filled: they only
// influence output columns >= N, which are never stored.
template <int EP>
__global__ void __launch_bounds__(256) kgemm(const u16* __restrict__ A,
                                             const u16* __restrict__ Bt,
                                             const float* __restrict__ bias,
                                             void* __restrict__ Cout,
                                             int M, int N, int K) {
  __shared__ __align__(16) u16 As[2][128 * 32];
  __shared__ __align__(16) u16 Bs[2][128 * 32];
  const int t = threadIdx.x;
  const int bm = blockIdx.y * 128, bn = blockIdx.x * 128;
  const int w = t >> 5;
  const int moff = (w >> 2) * 64;  // 2 wave rows
  const int noff = (w & 3) * 32;   // 4 wave cols

  auto load_stage = [&](int k0, int buf) {
    for (int i = t; i < 512; i += 256) {  // 128 rows x 4 uint4 each for A, B
      int r = i >> 2, c = (i & 3) * 8;
      g2lds_b128(&As[buf][r * 32 + c],
                 &A[(size_t)(bm + r) * K + k0 + c]);
      int rb = bn + r;
      rb = (rb < N) ? rb : (N - 1);  // clamp (see note above)
      g2lds_b128(&Bs[buf][r * 32 + c],
                 &Bt[(size_t)rb * K + k0 + c]);
    }
  };

  v8f acc[4][2];
  for (int mt = 0; mt < 4; ++mt)
    for (int nt = 0; nt < 2; ++nt) acc[mt][nt] = v8f_zero();

  load_stage(0, 0);
  wait_async();
  __syncthreads();

  int cur = 0;
  for (int k0 = 0; k0 < K; k0 += 32) {
    if (k0 + 32 < K) load_stage(k0 + 32, cur ^ 1);  // async, overlaps WMMAs

    FragAB af[4];
#pragma unroll
    for (int mt = 0; mt < 4; ++mt)
      af[mt] = ldfrag(As[cur], moff + mt * 16, 32, 0);
#pragma unroll
    for (int nt = 0; nt < 2; ++nt) {
      FragAB bf = ldfrag(Bs[cur], noff + nt * 16, 32, 0);
#pragma unroll
      for (int mt = 0; mt < 4; ++mt) acc[mt][nt] = wmma_bf16(af[mt], bf, acc[mt][nt]);
    }

    wait_async();     // next stage landed in LDS
    __syncthreads();  // all waves done reading cur / writing next
    cur ^= 1;
  }

  const int lr = t & 15, hig = (t >> 4) & 1;
#pragma unroll
  for (int mt = 0; mt < 4; ++mt)
#pragma unroll
    for (int nt = 0; nt < 2; ++nt)
#pragma unroll
      for (int i = 0; i < 8; ++i) {
        int row = bm + moff + mt * 16 + i + hig * 8;
        int col = bn + noff + nt * 16 + lr;
        if (col >= N) continue;
        float v = acc[mt][nt][i] + bias[col];
        if (EP == EP_F32) {
          ((float*)Cout)[(size_t)row * N + col] = v;
        } else if (EP == EP_BF16) {
          ((u16*)Cout)[(size_t)row * N + col] = f2bf(v);
        } else if (EP == EP_GELU_BF16) {
          float g = 0.5f * v * (1.f + erff(v * 0.70710678118654752f));
          ((u16*)Cout)[(size_t)row * N + col] = f2bf(g);
        } else {  // EP_HEADS_BF16: [b,s,h*ph] -> [b,h,s,ph]
          int b = row >> 9, q = row & (S_ - 1);
          int h = col >> 5, d = col & (PH_ - 1);
          ((u16*)Cout)[(((size_t)b * H_ + h) * S_ + q) * PH_ + d] = f2bf(v);
        }
      }
}

// ---------------------------------------------------- scores = Q @ K^T
// grid (ktile=4, qtile=4, b*h=200); 128x128 score tile, one k-step (d=32).
__global__ void __launch_bounds__(256) kqk(const u16* __restrict__ Qb,
                                           const u16* __restrict__ Kb,
                                           float* __restrict__ scores) {
  __shared__ __align__(16) u16 Qt[128 * 32];
  __shared__ __align__(16) u16 Kt[128 * 32];
  const int t = threadIdx.x;
  const int bh = blockIdx.z;
  const int qoff = blockIdx.y * 128;
  const int koff = blockIdx.x * 128;
  for (int i = t; i < 512; i += 256) {
    int r = i >> 2, c = (i & 3) * 8;
    g2lds_b128(&Qt[r * 32 + c], &Qb[((size_t)bh * S_ + qoff + r) * PH_ + c]);
    g2lds_b128(&Kt[r * 32 + c], &Kb[((size_t)bh * S_ + koff + r) * PH_ + c]);
  }
  wait_async();
  __syncthreads();
  const int w = t >> 5;
  const int wm = (w >> 1) * 32;  // 4 wave rows x 32
  const int wn = (w & 1) * 64;   // 2 wave cols x 64
  FragAB af[2];
#pragma unroll
  for (int mt = 0; mt < 2; ++mt) af[mt] = ldfrag(Qt, wm + mt * 16, 32, 0);
  const int lr = t & 15, hig = (t >> 4) & 1;
#pragma unroll
  for (int nt = 0; nt < 4; ++nt) {
    FragAB bf = ldfrag(Kt, wn + nt * 16, 32, 0);
#pragma unroll
    for (int mt = 0; mt < 2; ++mt) {
      v8f c = wmma_bf16(af[mt], bf, v8f_zero());
#pragma unroll
      for (int i = 0; i < 8; ++i) {
        int q = qoff + wm + mt * 16 + i + hig * 8;
        int k = koff + wn + nt * 16 + lr;
        scores[((size_t)bh * S_ + q) * S_ + k] = c[i];
      }
    }
  }
}

// ----------------------- rel-K bias + softmax, one workgroup per (b,q)
// R[h,k] = Q_bq[50,32] @ relk_gather[512,32]^T via WMMA; combined with
// QK^T scores; softmax over k kept in accumulators + shuffle reductions.
__global__ void __launch_bounds__(256) krelk_softmax(const u16* __restrict__ Qb,
                                                     const float* __restrict__ scores,
                                                     const float* __restrict__ rel_k_emb,
                                                     const int* __restrict__ relmat,
                                                     u16* __restrict__ probs) {
  __shared__ u16 emb[NR_ * PH_];
  __shared__ __align__(16) u16 Qh[64 * PH_];
  __shared__ int rm[128];
  __shared__ __align__(16) u16 Btl[128 * PH_];
  __shared__ float red[64 * 8];
  __shared__ float rowmax[64];
  __shared__ float rowsum[64];

  const int t = threadIdx.x;
  const int b = blockIdx.x >> 9;
  const int q = blockIdx.x & (S_ - 1);

  for (int i = t; i < NR_ * PH_; i += 256) emb[i] = f2bf(rel_k_emb[i]);
  for (int i = t; i < 64 * PH_; i += 256) {
    int h = i >> 5, d = i & 31;
    Qh[i] = (h < H_) ? Qb[(((size_t)b * H_ + h) * S_ + q) * PH_ + d] : (u16)0;
  }
  __syncthreads();

  const int w = t >> 5;
  FragAB af[4];
#pragma unroll
  for (int mt = 0; mt < 4; ++mt) af[mt] = ldfrag(Qh, mt * 16, 32, 0);

  v8f ct[4][4];
  for (int kt = 0; kt < 4; ++kt)
    for (int mt = 0; mt < 4; ++mt) ct[kt][mt] = v8f_zero();

  for (int kt = 0; kt < 4; ++kt) {
    const int koff = kt * 128;
    if (t < 128) rm[t] = relmat[((size_t)b * S_ + q) * S_ + koff + t];
    __syncthreads();
    for (int i = t; i < 128 * PH_; i += 256) {
      int j = i >> 5, d = i & 31;
      Btl[i] = emb[rm[j] * PH_ + d];
    }
    __syncthreads();
    FragAB bf = ldfrag(Btl, w * 16, 32, 0);  // wave w owns 16 key columns
#pragma unroll
    for (int mt = 0; mt < 4; ++mt) ct[kt][mt] = wmma_bf16(af[mt], bf, ct[kt][mt]);
    __syncthreads();
  }

  const int lr = t & 15, hig = (t >> 4) & 1;
  const float scale = 0.17677669529663689f;  // 1/sqrt(32)
#pragma unroll
  for (int kt = 0; kt < 4; ++kt)
    for (int mt = 0; mt < 4; ++mt)
      for (int i = 0; i < 8; ++i) {
        int row = mt * 16 + i + hig * 8;
        int col = kt * 128 + w * 16 + lr;
        float sv = ct[kt][mt][i];
        if (row < H_)
          sv = (sv + scores[(((size_t)b * H_ + row) * S_ + q) * S_ + col]) * scale;
        else
          sv = -1e30f;
        ct[kt][mt][i] = sv;
      }
  // row max: lane-local over 4 kt cols, shuffle over 16 lanes, LDS over 8 waves
  for (int mt = 0; mt < 4; ++mt)
    for (int i = 0; i < 8; ++i) {
      float mx = ct[0][mt][i];
      for (int kt = 1; kt < 4; ++kt) mx = fmaxf(mx, ct[kt][mt][i]);
      for (int off = 1; off < 16; off <<= 1) mx = fmaxf(mx, __shfl_xor(mx, off, 32));
      if (lr == 0) red[(mt * 16 + i + hig * 8) * 8 + w] = mx;
    }
  __syncthreads();
  if (t < 64) {
    float m = red[t * 8];
    for (int j = 1; j < 8; ++j) m = fmaxf(m, red[t * 8 + j]);
    rowmax[t] = m;
  }
  __syncthreads();
  for (int mt = 0; mt < 4; ++mt)
    for (int i = 0; i < 8; ++i) {
      int row = mt * 16 + i + hig * 8;
      float rmx = rowmax[row];
      float s = 0.f;
      for (int kt = 0; kt < 4; ++kt) s += __expf(ct[kt][mt][i] - rmx);
      for (int off = 1; off < 16; off <<= 1) s += __shfl_xor(s, off, 32);
      if (lr == 0) red[row * 8 + w] = s;
    }
  __syncthreads();
  if (t < 64) {
    float s = 0.f;
    for (int j = 0; j < 8; ++j) s += red[t * 8 + j];
    rowsum[t] = s;
  }
  __syncthreads();
  for (int kt = 0; kt < 4; ++kt)
    for (int mt = 0; mt < 4; ++mt)
      for (int i = 0; i < 8; ++i) {
        int row = mt * 16 + i + hig * 8;
        if (row >= H_) continue;
        int col = kt * 128 + w * 16 + lr;
        float p = __expf(ct[kt][mt][i] - rowmax[row]) / rowsum[row];
        probs[(((size_t)b * H_ + row) * S_ + q) * S_ + col] = f2bf(p);
      }
}

// ------------------------------------------------ ctx = probs @ V
// grid (qtile=4, b*h=200); each wave handles 16 q rows x 32 d.
__global__ void __launch_bounds__(256) kpv(const u16* __restrict__ probs,
                                           const u16* __restrict__ Vb,
                                           float* __restrict__ ctx) {
  __shared__ __align__(16) u16 Pt[128 * 32];
  __shared__ __align__(16) u16 Vt[32 * 32];
  const int t = threadIdx.x;
  const int bh = blockIdx.y;
  const int b = bh / H_, h = bh % H_;
  const int qoff = blockIdx.x * 128;
  const int w = t >> 5;
  v8f acc[2] = {v8f_zero(), v8f_zero()};
  for (int ko = 0; ko < 16; ++ko) {
    const int koff = ko * 32;
    for (int i = t; i < 512; i += 256) {
      int r = i >> 2, c = (i & 3) * 8;
      g2lds_b128(&Pt[r * 32 + c],
                 &probs[((size_t)bh * S_ + qoff + r) * S_ + koff + c]);
    }
    for (int i = t; i < 1024; i += 256) {  // Vt[d][k] = V[k][d] transpose
      int n = i >> 5, k = i & 31;
      Vt[n * 32 + k] = Vb[((size_t)bh * S_ + koff + k) * PH_ + n];
    }
    wait_async();
    __syncthreads();
    FragAB af = ldfrag(Pt, w * 16, 32, 0);
#pragma unroll
    for (int nt = 0; nt < 2; ++nt) {
      FragAB bf = ldfrag(Vt, nt * 16, 32, 0);
      acc[nt] = wmma_bf16(af, bf, acc[nt]);
    }
    __syncthreads();
  }
  const int lr = t & 15, hig = (t >> 4) & 1;
#pragma unroll
  for (int nt = 0; nt < 2; ++nt)
    for (int i = 0; i < 8; ++i) {
      int q = qoff + w * 16 + i + hig * 8;
      int d = nt * 16 + lr;
      ctx[((size_t)b * S_ + q) * D_ + h * PH_ + d] = acc[nt][i];
    }
}

// ------------------------ ctx += probs_bq[50,512] @ relv_gather[512,32]
// one workgroup per (b,q); 8 waves cover the 64x32 output (mt x nt).
__global__ void __launch_bounds__(256) krelv(const u16* __restrict__ probs,
                                             const float* __restrict__ rel_v_emb,
                                             const int* __restrict__ relmat,
                                             float* __restrict__ ctx) {
  __shared__ u16 emb[NR_ * PH_];
  __shared__ int rm[128];
  __shared__ __align__(16) u16 Pt[64 * 128];
  __shared__ __align__(16) u16 Btl[32 * 128];
  const int t = threadIdx.x;
  const int b = blockIdx.x >> 9;
  const int q = blockIdx.x & (S_ - 1);
  for (int i = t; i < NR_ * PH_; i += 256) emb[i] = f2bf(rel_v_emb[i]);
  const int w = t >> 5;
  const int mt = w >> 1, nt = w & 1;
  v8f acc = v8f_zero();
  for (int kt = 0; kt < 4; ++kt) {
    const int koff = kt * 128;
    if (t < 128) rm[t] = relmat[((size_t)b * S_ + q) * S_ + koff + t];
    for (int i = t; i < 64 * 16; i += 256) {  // 64 rows x 16 uint4
      int h = i >> 4, c = (i & 15) * 8;
      int hs = (h < H_) ? h : (H_ - 1);  // clamp; rows >= 50 never stored
      g2lds_b128(&Pt[h * 128 + c],
                 &probs[(((size_t)b * H_ + hs) * S_ + q) * S_ + koff + c]);
    }
    wait_async();
    __syncthreads();
    for (int i = t; i < 32 * 128; i += 256) {
      int n = i >> 7, k = i & 127;
      Btl[i] = emb[rm[k] * PH_ + n];
    }
    __syncthreads();
#pragma unroll
    for (int ks = 0; ks < 4; ++ks) {
      FragAB af = ldfrag(Pt, mt * 16, 128, ks * 32);
      FragAB bf = ldfrag(Btl, nt * 16, 128, ks * 32);
      acc = wmma_bf16(af, bf, acc);
    }
    __syncthreads();
  }
  const int lr = t & 15, hig = (t >> 4) & 1;
#pragma unroll
  for (int i = 0; i < 8; ++i) {
    int h = mt * 16 + i + hig * 8;
    if (h < H_) {
      size_t idx = ((size_t)b * S_ + q) * D_ + h * PH_ + nt * 16 + lr;
      ctx[idx] += acc[i];  // disjoint (b,q) per WG; ordered after kpv on stream
    }
  }
}

// ---------------------------------------------------------------- host
extern "C" void kernel_launch(void* const* d_in, const int* in_sizes, int n_in,
                              void* d_out, int out_size, void* d_ws, size_t ws_size,
                              hipStream_t stream) {
  (void)in_sizes; (void)n_in; (void)out_size; (void)ws_size;
  const float* hidden = (const float*)d_in[0];
  const float* Wq = (const float*)d_in[1];
  const float* bq = (const float*)d_in[2];
  const float* Wk = (const float*)d_in[3];
  const float* bk = (const float*)d_in[4];
  const float* Wv = (const float*)d_in[5];
  const float* bv = (const float*)d_in[6];
  const float* Wo = (const float*)d_in[7];
  const float* bo = (const float*)d_in[8];
  const float* rel_k = (const float*)d_in[9];
  const float* rel_v = (const float*)d_in[10];
  const float* W1 = (const float*)d_in[11];
  const float* b1 = (const float*)d_in[12];
  const float* W2 = (const float*)d_in[13];
  const float* b2 = (const float*)d_in[14];
  const int* relmat = (const int*)d_in[15];
  float* out = (float*)d_out;

  char* ws = (char*)d_ws;
  size_t off = 0;
  auto take = [&](size_t bytes) -> char* {
    char* p = ws + off;
    off += (bytes + 255) & ~(size_t)255;
    return p;
  };
  // ~405 MB total workspace
  u16* hbf  = (u16*)take((size_t)MS_ * D_ * 2);
  u16* WqT  = (u16*)take((size_t)D_ * D_ * 2);
  u16* WkT  = (u16*)take((size_t)D_ * D_ * 2);
  u16* WvT  = (u16*)take((size_t)D_ * D_ * 2);
  u16* WoT  = (u16*)take((size_t)D_ * D_ * 2);
  u16* W1T  = (u16*)take((size_t)D_ * D_ * 2);
  u16* W2T  = (u16*)take((size_t)D_ * D_ * 2);
  u16* Qb   = (u16*)take((size_t)MS_ * D_ * 2);
  u16* Kb   = (u16*)take((size_t)MS_ * D_ * 2);
  u16* Vb   = (u16*)take((size_t)MS_ * D_ * 2);
  float* sc = (float*)take((size_t)B_ * H_ * S_ * S_ * 4);
  u16* pr   = (u16*)take((size_t)B_ * H_ * S_ * S_ * 2);
  float* ctx = (float*)take((size_t)MS_ * D_ * 4);
  u16* ctxb = (u16*)take((size_t)MS_ * D_ * 2);
  u16* attnb = (u16*)take((size_t)MS_ * D_ * 2);
  u16* h1b  = (u16*)take((size_t)MS_ * D_ * 2);

  const dim3 blk(256);
  const int nelem = MS_ * D_;
  const dim3 gcast((nelem + 255) / 256);
  const dim3 gtr(50, 50);
  const dim3 ggemm((D_ + 127) / 128, MS_ / 128);  // (13, 16)

  kcast<<<gcast, blk, 0, stream>>>(hidden, hbf, nelem);
  ktranspose_cast<<<gtr, blk, 0, stream>>>(Wq, WqT, D_, D_);
  ktranspose_cast<<<gtr, blk, 0, stream>>>(Wk, WkT, D_, D_);
  ktranspose_cast<<<gtr, blk, 0, stream>>>(Wv, WvT, D_, D_);
  ktranspose_cast<<<gtr, blk, 0, stream>>>(Wo, WoT, D_, D_);
  ktranspose_cast<<<gtr, blk, 0, stream>>>(W1, W1T, D_, D_);
  ktranspose_cast<<<gtr, blk, 0, stream>>>(W2, W2T, D_, D_);

  kgemm<EP_HEADS_BF16><<<ggemm, blk, 0, stream>>>(hbf, WqT, bq, Qb, MS_, D_, D_);
  kgemm<EP_HEADS_BF16><<<ggemm, blk, 0, stream>>>(hbf, WkT, bk, Kb, MS_, D_, D_);
  kgemm<EP_HEADS_BF16><<<ggemm, blk, 0, stream>>>(hbf, WvT, bv, Vb, MS_, D_, D_);

  kqk<<<dim3(S_ / 128, S_ / 128, B_ * H_), blk, 0, stream>>>(Qb, Kb, sc);
  krelk_softmax<<<dim3(B_ * S_), blk, 0, stream>>>(Qb, sc, rel_k, relmat, pr);
  kpv<<<dim3(S_ / 128, B_ * H_), blk, 0, stream>>>(pr, Vb, ctx);
  krelv<<<dim3(B_ * S_), blk, 0, stream>>>(pr, rel_v, relmat, ctx);

  kcast<<<gcast, blk, 0, stream>>>(ctx, ctxb, nelem);
  kgemm<EP_BF16><<<ggemm, blk, 0, stream>>>(ctxb, WoT, bo, attnb, MS_, D_, D_);
  kgemm<EP_GELU_BF16><<<ggemm, blk, 0, stream>>>(attnb, W1T, b1, h1b, MS_, D_, D_);
  kgemm<EP_F32><<<ggemm, blk, 0, stream>>>(h1b, W2T, b2, out, MS_, D_, D_);
}
